// SubmanifoldConvolution_56143812493551
// MI455X (gfx1250) — matrix operand order repaired and checked by reading, not compiled, not run on hardware
//
#include <hip/hip_runtime.h>

typedef __attribute__((ext_vector_type(16))) __bf16       v16bf;
typedef __attribute__((ext_vector_type(8)))  float        v8f;
typedef __attribute__((ext_vector_type(4)))  unsigned int v4u;

#define NSITES 100000
#define CIN    32
#define COUT   32
#define KOFF   27
#define FRAG_ELEMS (KOFF * 2 * 32 * 16)   // 27648 bf16 elements per (hi|lo) weight array

// ---------------------------------------------------------------------------
// Pre-kernel 1: repack weight[k][cin][cout] (f32) into per-lane WMMA B-fragment
// order, split into bf16 hi + bf16 lo residual.
// Fragment element e = ((k*2 + ch)*32 + lane)*16 + j maps to
//   K-row (cin) = (j/8)*16 + (lane/16)*8 + j%8,  column (cout) = (lane%16) + 16*ch
// ---------------------------------------------------------------------------
__global__ __launch_bounds__(256) void prep_weights_kernel(
    const float* __restrict__ w, __bf16* __restrict__ whi, __bf16* __restrict__ wlo)
{
    int e = blockIdx.x * 256 + threadIdx.x;
    if (e >= FRAG_ELEMS) return;
    int j  = e & 15;
    int L  = (e >> 4) & 31;
    int ch = (e >> 9) & 1;
    int k  = e >> 10;
    int krow = ((j >> 3) << 4) + ((L >> 4) << 3) + (j & 7);          // cin (reduction) row
    int col  = (L & 15) + (ch << 4);         // cout column
    float v  = w[(k * CIN + krow) * COUT + col];
    __bf16 h = (__bf16)v;
    whi[e] = h;
    wlo[e] = (__bf16)(v - (float)h);
}

// ---------------------------------------------------------------------------
// Pre-kernel 2: split features once into a bf16 hi/lo image.
// Row n occupies one 128B line: [n*64 .. n*64+31] = hi, [n*64+32 .. +63] = lo.
// Row NSITES is all zeros: masked-out rulebook taps gather it for exact zeros.
// ---------------------------------------------------------------------------
__global__ __launch_bounds__(256) void prep_feats_kernel(
    const float* __restrict__ f, __bf16* __restrict__ fs)
{
    int e = blockIdx.x * 256 + threadIdx.x;   // e = n*32 + c
    if (e < 64) fs[NSITES * 64 + e] = (__bf16)0.0f;   // zero row for masked taps
    if (e >= NSITES * CIN) return;
    int n = e >> 5;
    int c = e & 31;
    float v  = f[e];
    __bf16 h = (__bf16)v;
    fs[n * 64 + c]      = h;
    fs[n * 64 + 32 + c] = (__bf16)(v - (float)h);
}

// ---------------------------------------------------------------------------
// Pre-kernel 3: fold the mask into the rulebook. Masked-out entries point at
// the zero feature row, removing the mask load + 16 v_and per inner iteration.
// ---------------------------------------------------------------------------
__global__ __launch_bounds__(256) void prep_idx_kernel(
    const int* __restrict__ idx, const unsigned char* __restrict__ mask,
    int* __restrict__ midx)
{
    int e = blockIdx.x * 256 + threadIdx.x;
    if (e >= NSITES * KOFF) return;
    midx[e] = mask[e] ? idx[e] : NSITES;
}

// ---------------------------------------------------------------------------
// Main kernel: one wave32 computes a 16-site x 32-Cout tile.
// Per offset k: gather pre-split, pre-masked bf16 fragments (one 128B line per
// neighbor) and issue 6x v_wmma_f32_16x16x32_bf16 (Ah*Bh + Al*Bh + Ah*Bl per
// Cout half). Pure load->WMMA pipeline, no per-tap conversion or masking VALU.
// ---------------------------------------------------------------------------
__global__ __launch_bounds__(256) void spconv_wmma_kernel(
    const __bf16* __restrict__ fsplit,
    const float*  __restrict__ bias,
    const int*    __restrict__ midx,
    const __bf16* __restrict__ whi,
    const __bf16* __restrict__ wlo,
    float*        __restrict__ out)
{
    const int lane = threadIdx.x & 31;
    const int wave = threadIdx.x >> 5;
    const int tile = blockIdx.x * 8 + wave;
    if (tile * 16 >= NSITES) return;          // wave-uniform: EXEC stays all-ones

    const int m       = lane & 15;            // site row in tile / B column
    const int half    = lane >> 4;            // lane half selects A K-interleave
    const int site    = tile * 16 + m;
    const int rb      = site * KOFF;          // rulebook base
    const int cinbase = half ? 8 : 0;         // A layout: K = cinbase+0..7, 16+cinbase+0..7

    v8f c0 = {};
    v8f c1 = {};

    union AV { v4u u[2]; v16bf v; };

#pragma unroll 3
    for (int k = 0; k < KOFF; ++k) {
        const int nb = midx[rb + k];

        // Gather this lane's A fragments from the pre-split 128B row (b128 x4).
        const __bf16* src = fsplit + nb * 64 + cinbase;
        AV ah, al;
        ah.u[0] = *(const v4u*)(src);         // hi: cin base..base+7
        ah.u[1] = *(const v4u*)(src + 16);    // hi: cin 16+base..+7
        al.u[0] = *(const v4u*)(src + 32);    // lo: cin base..base+7
        al.u[1] = *(const v4u*)(src + 48);    // lo: cin 16+base..+7

        // B fragments (pre-transposed, L2/L0 resident): 32B per fragment/lane.
        const int fb0 = ((k * 2 + 0) * 32 + lane) * 16;
        const int fb1 = ((k * 2 + 1) * 32 + lane) * 16;
        const v16bf bh0 = *(const v16bf*)(whi + fb0);
        const v16bf bl0 = *(const v16bf*)(wlo + fb0);
        const v16bf bh1 = *(const v16bf*)(whi + fb1);
        const v16bf bl1 = *(const v16bf*)(wlo + fb1);

        // D = A x B + C, three-term split product per Cout half.
        c0 = __builtin_amdgcn_wmma_f32_16x16x32_bf16(false, ah.v, false, bh0, (short)0, c0, false, false);
        c1 = __builtin_amdgcn_wmma_f32_16x16x32_bf16(false, ah.v, false, bh1, (short)0, c1, false, false);
        c0 = __builtin_amdgcn_wmma_f32_16x16x32_bf16(false, al.v, false, bh0, (short)0, c0, false, false);
        c1 = __builtin_amdgcn_wmma_f32_16x16x32_bf16(false, al.v, false, bh1, (short)0, c1, false, false);
        c0 = __builtin_amdgcn_wmma_f32_16x16x32_bf16(false, ah.v, false, bl0, (short)0, c0, false, false);
        c1 = __builtin_amdgcn_wmma_f32_16x16x32_bf16(false, ah.v, false, bl1, (short)0, c1, false, false);
    }

    // C/D layout: VGPR r holds M = r + 8*(lane/16), N = lane%16.
    const float b0 = bias[m];
    const float b1 = bias[m + 16];
#pragma unroll
    for (int r = 0; r < 8; ++r) {
        const int row = tile * 16 + r + half * 8;
        out[row * COUT + m]      = c0[r] + b0;
        out[row * COUT + 16 + m] = c1[r] + b1;
    }
}

// ---------------------------------------------------------------------------
// Inputs (setup_inputs order):
//   d_in[0] features  f32 [N,32]
//   d_in[1] weight    f32 [27,32,32]
//   d_in[2] bias      f32 [32]
//   d_in[3] nbr_idx   i32 [N,27]
//   d_in[4] nbr_mask  bool [N,27]
// Output: f32 [N,32]
// Workspace: whi[27648] bf16 | wlo[27648] bf16 | fsplit[(N+1)*64] bf16 | midx[N*27] i32
// ---------------------------------------------------------------------------
extern "C" void kernel_launch(void* const* d_in, const int* in_sizes, int n_in,
                              void* d_out, int out_size, void* d_ws, size_t ws_size,
                              hipStream_t stream) {
    (void)in_sizes; (void)n_in; (void)out_size; (void)ws_size;

    const float*         feats    = (const float*)d_in[0];
    const float*         weight   = (const float*)d_in[1];
    const float*         bias     = (const float*)d_in[2];
    const int*           nbr_idx  = (const int*)d_in[3];
    const unsigned char* nbr_mask = (const unsigned char*)d_in[4];

    __bf16* whi    = (__bf16*)d_ws;
    __bf16* wlo    = whi + FRAG_ELEMS;
    __bf16* fsplit = wlo + FRAG_ELEMS;                       // +110,592B (128B aligned)
    int*    midx   = (int*)(fsplit + (NSITES + 1) * 64);     // +12,800,128B (4B aligned)

    prep_weights_kernel<<<(FRAG_ELEMS + 255) / 256, 256, 0, stream>>>(weight, whi, wlo);
    prep_feats_kernel<<<(NSITES * CIN + 255) / 256, 256, 0, stream>>>(feats, fsplit);
    prep_idx_kernel<<<(NSITES * KOFF + 255) / 256, 256, 0, stream>>>(nbr_idx, nbr_mask, midx);

    const int ntiles  = (NSITES + 15) / 16;   // 6250 (N divisible by 16 -> all tiles full)
    const int nblocks = (ntiles + 7) / 8;     // 782 blocks x 8 waves
    spconv_wmma_kernel<<<nblocks, 256, 0, stream>>>(fsplit, bias, midx,
                                                    whi, wlo, (float*)d_out);
}
